// MyMSA_120259085167
// MI455X (gfx1250) — matrix operand order repaired and checked by reading, not compiled, not run on hardware
//
#include <hip/hip_runtime.h>

#define BB 16
#define SS 1024
#define DD 1024
#define HH 16
#define DHD 64
#define LSTR 72   // padded LDS row stride (halfwords); 144B rows keep b128 frag
                  // loads 16B-aligned and bank-conflict-free across a 16-lane group

typedef __attribute__((ext_vector_type(16))) _Float16 v16h;
typedef __attribute__((ext_vector_type(8)))  _Float16 v8h;
typedef __attribute__((ext_vector_type(8)))  float    v8f;

#if __has_builtin(__builtin_amdgcn_tensor_load_to_lds)
#define USE_TDM 1
typedef __attribute__((ext_vector_type(4))) unsigned int u32x4;
typedef __attribute__((ext_vector_type(8))) int          i32x8;
typedef __attribute__((ext_vector_type(4))) int          i32x4;

// One TDM descriptor: DMA a 64x64 f16 tile (row-major, row stride 64 elems in
// global) into LDS with 16B padding appended to every 128B row -> LSTR=72.
static __device__ __forceinline__ void tdm_load_tile_64x64_f16(
    const _Float16* gsrc, _Float16* ldsDst) {
  unsigned long long ga = (unsigned long long)(uintptr_t)gsrc;
  unsigned int lds = (unsigned int)(uintptr_t)ldsDst;   // low 32b = LDS byte offset
  u32x4 g0 = {
      1u,                                               // count=1 (valid D#)
      lds,                                              // lds_addr
      (unsigned int)(ga & 0xffffffffu),                 // global_addr[31:0]
      (unsigned int)((ga >> 32) & 0x1ffffffu) | 0x80000000u  // [56:32] | type=2
  };
  i32x8 g1 = {
      (1 << 16)          // data_size = 2 bytes
    | (1 << 20)          // pad_enable
    | (4 << 22)          // pad_interval: pad after 32 DWORDs (one 128B row)
    | (3 << 25),         // pad_amount: 4 DWORDs (16B)  -> 144B LDS row stride
      (64 << 16),        // [47:32] atomic_barrier=0 ; [63:48] tensor_dim0.lo = 64
      (64 << 16),        // [79:64] tensor_dim0.hi = 0 ; [95:80].. tensor_dim1.lo = 64
      (64 << 16),        // tensor_dim1.hi = 0 ; [127:112] tile_dim0 = 64
      64,                // [143:128] tile_dim1 = 64 ; tile_dim2 = 0
      64,                // tensor_dim0_stride.lo = 64 elements
      0,                 // stride hi / tensor_dim1_stride.lo
      0                  // tensor_dim1_stride.hi
  };
  i32x4 z4 = {0, 0, 0, 0};
#if defined(__clang_major__) && (__clang_major__ >= 23)
  i32x8 z8 = {0, 0, 0, 0, 0, 0, 0, 0};
  __builtin_amdgcn_tensor_load_to_lds(g0, g1, z4, z4, z8, 0);
#else
  __builtin_amdgcn_tensor_load_to_lds(g0, g1, z4, z4, 0);
#endif
}
#endif  // USE_TDM

static __device__ __forceinline__ v8f wmma_f16(v16h a, v16h b, v8f c) {
  // D = A(16x32) * B(32x16) + C, f32 accumulate
  return __builtin_amdgcn_wmma_f32_16x16x32_f16(false, a, false, b, (short)0, c, false, false);
}

static __device__ __forceinline__ v16h cat8(v8h lo, v8h hi) {
  return __builtin_shufflevector(lo, hi, 0, 1, 2, 3, 4, 5, 6, 7,
                                         8, 9, 10, 11, 12, 13, 14, 15);
}

// A fragment: 16x32 f16, storage [M][K] row-major with given stride.
// Lane holds row m0+l16 -> two contiguous 16-byte chunks per lane.
static __device__ __forceinline__ v16h load_a_frag(const _Float16* src, int stride,
                                                   int m0, int k0, int l16, int half) {
  const _Float16* p = src + (size_t)(m0 + l16) * stride + k0 + half * 16;
  v8h lo = *(const v8h*)(p);
  v8h hi = *(const v8h*)(p + 8);
  return cat8(lo, hi);
}

// B fragment: 32x16 f16, element (k,n) stored at src[n*stride + k].
// Lane holds column n0+l16, K = k0+16*half + 0..15 -> 32 contiguous bytes.
static __device__ __forceinline__ v16h load_b_frag_nk(const _Float16* src, int stride,
                                                      int k0, int n0, int l16, int half) {
  const _Float16* p = src + (size_t)(n0 + l16) * stride + k0 + half * 16;
  v8h lo = *(const v8h*)(p);
  v8h hi = *(const v8h*)(p + 8);
  return cat8(lo, hi);
}

// Online-softmax update for one 16-row strip: scales, reduces across the
// 16-lane row group, rescales running state, writes P (f16) to pRows.
static __device__ __forceinline__ void softmax_update(
    v8f (&sacc)[4], float (&m_i)[8], float (&l_i)[8], v8f (&oacc)[4],
    _Float16* pRows, int l16, int half, float scale) {
#pragma unroll
  for (int i = 0; i < 8; ++i) {
    float rm = fmaxf(fmaxf(sacc[0][i], sacc[1][i]),
                     fmaxf(sacc[2][i], sacc[3][i])) * scale;
    rm = fmaxf(rm, __shfl_xor(rm, 1, 32));
    rm = fmaxf(rm, __shfl_xor(rm, 2, 32));
    rm = fmaxf(rm, __shfl_xor(rm, 4, 32));
    rm = fmaxf(rm, __shfl_xor(rm, 8, 32));
    float mnew  = fmaxf(m_i[i], rm);
    float alpha = __expf(m_i[i] - mnew);
    m_i[i] = mnew;
    float rs = 0.f;
#pragma unroll
    for (int t = 0; t < 4; ++t) {
      float p = __expf(sacc[t][i] * scale - mnew);
      rs += p;
      pRows[(half * 8 + i) * LSTR + t * 16 + l16] = (_Float16)p;
    }
    rs += __shfl_xor(rs, 1, 32);
    rs += __shfl_xor(rs, 2, 32);
    rs += __shfl_xor(rs, 4, 32);
    rs += __shfl_xor(rs, 8, 32);
    l_i[i] = l_i[i] * alpha + rs;
#pragma unroll
    for (int t = 0; t < 4; ++t) oacc[t][i] *= alpha;
  }
}

// ---------------------------------------------------------------------------
// Kernel 1: per-head QKV projection.  q = x @ W + b, f16 out to workspace.
// grid(S/64, H, B), block 128 (4 waves, each owning a 16-row strip).
// Weights are staged TRANSPOSED ([out][in]) so B-fragments are contiguous.
// ---------------------------------------------------------------------------
__global__ __launch_bounds__(128) void qkv_proj_kernel(
    const float* __restrict__ seq,
    const float* __restrict__ Wq, const float* __restrict__ bq,
    const float* __restrict__ Wk, const float* __restrict__ bk,
    const float* __restrict__ Wv, const float* __restrict__ bv,
    _Float16* __restrict__ qbuf, _Float16* __restrict__ kbuf, _Float16* __restrict__ vbuf) {
  __shared__ __align__(16) _Float16 xLds[64 * LSTR];
  __shared__ __align__(16) _Float16 wtLds[3][64 * LSTR];   // [out e][in d]

  const int tid  = threadIdx.x;
  const int wave = tid >> 5;
  const int lane = tid & 31;
  const int l16  = lane & 15;
  const int half = lane >> 4;

  const int rb = blockIdx.x;   // 64-row chunk of the sequence
  const int h  = blockIdx.y;
  const int b  = blockIdx.z;
  const int bh = b * HH + h;

  // Stage x tile [64 x 64] (fp32 -> f16), rows strided by D in global.
  const float* xg = seq + ((size_t)b * SS + (size_t)rb * 64) * DD + h * DHD;
  for (int c = tid; c < 64 * 16; c += 128) {
    int r = c >> 4, off = (c & 15) * 4;
    float4 f = *(const float4*)(xg + (size_t)r * DD + off);
    _Float16* d = xLds + r * LSTR + off;
    d[0] = (_Float16)f.x; d[1] = (_Float16)f.y;
    d[2] = (_Float16)f.z; d[3] = (_Float16)f.w;
  }
  // Stage W transposed: element (d_in, e_out) -> wtLds[e*LSTR + d]
  const float* wsrc[3] = {Wq + (size_t)h * DHD * DHD,
                          Wk + (size_t)h * DHD * DHD,
                          Wv + (size_t)h * DHD * DHD};
#pragma unroll
  for (int m = 0; m < 3; ++m) {
    const float* wg = wsrc[m];
    for (int c = tid; c < 64 * 16; c += 128) {
      int d_in = c >> 4, e0 = (c & 15) * 4;
      float4 f = *(const float4*)(wg + d_in * 64 + e0);
      _Float16* dst = wtLds[m];
      dst[(e0 + 0) * LSTR + d_in] = (_Float16)f.x;
      dst[(e0 + 1) * LSTR + d_in] = (_Float16)f.y;
      dst[(e0 + 2) * LSTR + d_in] = (_Float16)f.z;
      dst[(e0 + 3) * LSTR + d_in] = (_Float16)f.w;
    }
  }
  __syncthreads();

  v16h ax0 = load_a_frag(xLds, LSTR, wave * 16, 0,  l16, half);
  v16h ax1 = load_a_frag(xLds, LSTR, wave * 16, 32, l16, half);

  const float* bias[3] = {bq + h * DHD, bk + h * DHD, bv + h * DHD};
  _Float16* dsts[3] = {qbuf, kbuf, vbuf};
#pragma unroll
  for (int m = 0; m < 3; ++m) {
    _Float16* dst = dsts[m] + (size_t)bh * SS * DHD + (size_t)rb * 64 * DHD;
#pragma unroll
    for (int t = 0; t < 4; ++t) {
      float bb = bias[m][t * 16 + l16];   // bias is per output column
      v8f c;
#pragma unroll
      for (int i = 0; i < 8; ++i) c[i] = bb;
      v16h b0 = load_b_frag_nk(wtLds[m], LSTR, 0,  t * 16, l16, half);
      v16h b1 = load_b_frag_nk(wtLds[m], LSTR, 32, t * 16, l16, half);
      c = wmma_f16(ax0, b0, c);
      c = wmma_f16(ax1, b1, c);
#pragma unroll
      for (int i = 0; i < 8; ++i) {
        int row = wave * 16 + half * 8 + i;
        dst[(size_t)row * DHD + t * 16 + l16] = (_Float16)c[i];
      }
    }
  }
}

// ---------------------------------------------------------------------------
// Kernel 2: flash attention over one (b,h), 128-query block per workgroup.
// grid(S/128, H, B), block 128; each wave owns TWO 16-row query strips so
// every K/V B-fragment in registers feeds two WMMAs.
// K tile: TDM tensor_load_to_lds ([key][dh], padded rows, TENSORcnt).
// V tile: staged TRANSPOSED ([dh][key]) by the VALU path while TDM runs.
// ---------------------------------------------------------------------------
__global__ __launch_bounds__(128) void flash_attn_kernel(
    const _Float16* __restrict__ qbuf, const _Float16* __restrict__ kbuf,
    const _Float16* __restrict__ vbuf, float* __restrict__ out) {
  __shared__ __align__(16) _Float16 kLds[64 * LSTR];        // [key][dh]
  __shared__ __align__(16) _Float16 vtLds[64 * LSTR];       // [dh][key]
  __shared__ __align__(16) _Float16 pLds[4][32 * LSTR];     // wave-private P (2 strips)

  const int tid  = threadIdx.x;
  const int wave = tid >> 5;
  const int lane = tid & 31;
  const int l16  = lane & 15;
  const int half = lane >> 4;

  const int qb = blockIdx.x;     // 128-query chunk
  const int h  = blockIdx.y;
  const int b  = blockIdx.z;
  const int bh = b * HH + h;

  // Q fragments: two 16-row strips per wave, resident for the whole loop.
  const _Float16* qg = qbuf + (size_t)bh * SS * DHD + (size_t)qb * 128 * DHD;
  v16h aq[2][2];
#pragma unroll
  for (int s = 0; s < 2; ++s) {
    aq[s][0] = load_a_frag(qg, DHD, wave * 32 + s * 16, 0,  l16, half);
    aq[s][1] = load_a_frag(qg, DHD, wave * 32 + s * 16, 32, l16, half);
  }

  float m_i[2][8], l_i[2][8];
  v8f oacc[2][4];
#pragma unroll
  for (int s = 0; s < 2; ++s) {
#pragma unroll
    for (int i = 0; i < 8; ++i) { m_i[s][i] = -1e30f; l_i[s][i] = 0.f; }
#pragma unroll
    for (int t = 0; t < 4; ++t)
#pragma unroll
      for (int i = 0; i < 8; ++i) oacc[s][t][i] = 0.f;
  }

  const _Float16* kg0 = kbuf + (size_t)bh * SS * DHD;
  const _Float16* vg0 = vbuf + (size_t)bh * SS * DHD;
  const float scale = 0.125f;   // 1/sqrt(64)
  _Float16* pw = pLds[wave];

  for (int kt = 0; kt < SS / 64; ++kt) {
    __syncthreads();            // previous iteration's LDS consumers done
    const _Float16* kg = kg0 + (size_t)kt * 64 * DHD;
    const _Float16* vg = vg0 + (size_t)kt * 64 * DHD;

#ifdef USE_TDM
    if (wave == 0) tdm_load_tile_64x64_f16(kg, kLds);   // K tile via TDM DMA
#endif
    // V tile transpose-staged by all threads (overlaps the TDM transfer).
    for (int c = tid; c < 64 * 8; c += 128) {
      int r = c >> 3, off = (c & 7) * 8;     // key r, dh chunk off..off+7
#ifndef USE_TDM
      *(v8h*)(kLds + r * LSTR + off) = *(const v8h*)(kg + (size_t)r * DHD + off);
#endif
      v8h vv = *(const v8h*)(vg + (size_t)r * DHD + off);
#pragma unroll
      for (int j = 0; j < 8; ++j)            // transpose: vtLds[dh][key]
        vtLds[(off + j) * LSTR + r] = vv[j];
    }
    if (kt + 1 < SS / 64) {     // warm GL2 for the next V tile
      __builtin_prefetch(vg0 + (size_t)(kt + 1) * 64 * DHD, 0, 1);
    }
#ifdef USE_TDM
    if (wave == 0) __builtin_amdgcn_s_wait_tensorcnt(0);
#endif
    __syncthreads();

    // S = Q * K^T : each K B-fragment (in registers) feeds both strips.
    v8f sacc[2][4];
#pragma unroll
    for (int s = 0; s < 2; ++s)
#pragma unroll
      for (int t = 0; t < 4; ++t)
#pragma unroll
        for (int i = 0; i < 8; ++i) sacc[s][t][i] = 0.f;
#pragma unroll
    for (int t = 0; t < 4; ++t) {
      v16h bk0 = load_b_frag_nk(kLds, LSTR, 0,  t * 16, l16, half);
      v16h bk1 = load_b_frag_nk(kLds, LSTR, 32, t * 16, l16, half);
      sacc[0][t] = wmma_f16(aq[0][0], bk0, sacc[0][t]);
      sacc[0][t] = wmma_f16(aq[0][1], bk1, sacc[0][t]);
      sacc[1][t] = wmma_f16(aq[1][0], bk0, sacc[1][t]);
      sacc[1][t] = wmma_f16(aq[1][1], bk1, sacc[1][t]);
    }

    softmax_update(sacc[0], m_i[0], l_i[0], oacc[0], pw,             l16, half, scale);
    softmax_update(sacc[1], m_i[1], l_i[1], oacc[1], pw + 16 * LSTR, l16, half, scale);
    __builtin_amdgcn_wave_barrier();   // keep wave-private LDS store->load ordered

    // O += P * V : each V B-fragment (in registers) feeds both strips.
    v16h pa[2][2];
#pragma unroll
    for (int s = 0; s < 2; ++s) {
      pa[s][0] = load_a_frag(pw, LSTR, s * 16, 0,  l16, half);
      pa[s][1] = load_a_frag(pw, LSTR, s * 16, 32, l16, half);
    }
#pragma unroll
    for (int t = 0; t < 4; ++t) {
      v16h bv0 = load_b_frag_nk(vtLds, LSTR, 0,  t * 16, l16, half);
      v16h bv1 = load_b_frag_nk(vtLds, LSTR, 32, t * 16, l16, half);
      oacc[0][t] = wmma_f16(pa[0][0], bv0, oacc[0][t]);
      oacc[0][t] = wmma_f16(pa[0][1], bv1, oacc[0][t]);
      oacc[1][t] = wmma_f16(pa[1][0], bv0, oacc[1][t]);
      oacc[1][t] = wmma_f16(pa[1][1], bv1, oacc[1][t]);
    }
  }

  // Normalize and write out [B,S,D] fp32, head h occupying columns h*64..h*64+63.
  float* og = out + ((size_t)b * SS + (size_t)qb * 128) * DD + h * DHD;
#pragma unroll
  for (int s = 0; s < 2; ++s)
#pragma unroll
    for (int i = 0; i < 8; ++i) {
      float inv = 1.f / l_i[s][i];
      int row = wave * 32 + s * 16 + half * 8 + i;
#pragma unroll
      for (int t = 0; t < 4; ++t)
        og[(size_t)row * DD + t * 16 + l16] = oacc[s][t][i] * inv;
    }
}

extern "C" void kernel_launch(void* const* d_in, const int* in_sizes, int n_in,
                              void* d_out, int out_size, void* d_ws, size_t ws_size,
                              hipStream_t stream) {
  (void)in_sizes; (void)n_in; (void)out_size; (void)ws_size;
  const float* seq = (const float*)d_in[0];
  const float* Wq  = (const float*)d_in[1];
  const float* bq  = (const float*)d_in[2];
  const float* Wk  = (const float*)d_in[3];
  const float* bk  = (const float*)d_in[4];
  const float* Wv  = (const float*)d_in[5];
  const float* bv  = (const float*)d_in[6];
  float* outp = (float*)d_out;

  // Workspace: f16 Q, K, V as [B*H, S, 64]  (3 x 32 MB = 96 MB)
  _Float16* qbuf = (_Float16*)d_ws;
  size_t per = (size_t)BB * HH * SS * DHD;
  _Float16* kbuf = qbuf + per;
  _Float16* vbuf = kbuf + per;

  dim3 gridP(SS / 64, HH, BB), block(128);
  hipLaunchKernelGGL(qkv_proj_kernel, gridP, block, 0, stream,
                     seq, Wq, bq, Wk, bk, Wv, bv, qbuf, kbuf, vbuf);
  dim3 gridA(SS / 128, HH, BB);
  hipLaunchKernelGGL(flash_attn_kernel, gridA, block, 0, stream,
                     qbuf, kbuf, vbuf, outp);
}
